// Quantizer_73873437491354
// MI455X (gfx1250) — matrix operand order repaired and checked by reading, not compiled, not run on hardware
//
#include <hip/hip_runtime.h>

// ---------------------------------------------------------------------------
// Quantizer pipeline for MI455X (gfx1250): bf16 WMMA everywhere.
//   B=8 T=2048 D=1024 Dc=256 K=8192, M = B*T = 16384
// ---------------------------------------------------------------------------

typedef __attribute__((ext_vector_type(16))) __bf16 v16bf;
typedef __attribute__((ext_vector_type(8)))  __bf16 v8bf;
typedef __attribute__((ext_vector_type(8)))  float  v8f;
typedef __attribute__((ext_vector_type(4)))  int    v4i;

#define MROWS 16384
#define DHID  1024
#define DC    256
#define KCODE 8192

union AFrag { v16bf v; v8bf h[2]; };

__device__ __forceinline__ v8f wmma_bf16(v16bf a, v16bf b, v8f c) {
  // (neg_a, A, neg_b, B, c_mod, C, reuse_a, reuse_b)
  return __builtin_amdgcn_wmma_f32_16x16x32_bf16(false, a, false, b, (short)0, c,
                                                 false, false);
}

// ---------------------------------------------------------------------------
// Async global->LDS staging (CDNA5 GLOBAL_LOAD_ASYNC_TO_LDS_B128, ASYNCcnt).
// Builtin signature (from compiler diagnostics): param0 = v4i* in global
// (__device__/AS1) space; param1 = LDS-side pointer; then imm offset, imm cpol.
// Guarded: falls back to a VGPR-bounce copy if the builtins are absent.
// ---------------------------------------------------------------------------
#if __has_builtin(__builtin_amdgcn_global_load_async_to_lds_b128) && \
    __has_builtin(__builtin_amdgcn_s_wait_asynccnt)
#define USE_ASYNC_LDS 1
#else
#define USE_ASYNC_LDS 0
#endif

#if USE_ASYNC_LDS
typedef __attribute__((address_space(1))) v4i* gv4i_p;
typedef __attribute__((address_space(3))) v4i* lv4i_p;
#endif

__device__ __forceinline__ void copy32B_to_lds(const __bf16* g, __bf16* l) {
#if USE_ASYNC_LDS
  // IOFFSET is added to both the global and LDS addresses (ISA 10.x async ops)
  __builtin_amdgcn_global_load_async_to_lds_b128((gv4i_p)g, (lv4i_p)l, 0, 0);
  __builtin_amdgcn_global_load_async_to_lds_b128((gv4i_p)g, (lv4i_p)l, 16, 0);
#else
  ((int4*)l)[0] = ((const int4*)g)[0];
  ((int4*)l)[1] = ((const int4*)g)[1];
#endif
}

__device__ __forceinline__ void wait_async_lds() {
#if USE_ASYNC_LDS
  __builtin_amdgcn_s_wait_asynccnt(0);
#endif
}

// ---------------------------------------------------------------------------
// Kernel 1: normalize codebook rows -> f32 + bf16 copies; zero the loss slot.
// One wave per row (256 floats -> 8 per lane). 1024 blocks x 256 threads.
// ---------------------------------------------------------------------------
__global__ void __launch_bounds__(256)
prep_codebook(const float* __restrict__ cb, float* __restrict__ cn_f32,
              __bf16* __restrict__ cn_b, float* __restrict__ loss_slot) {
  if (blockIdx.x == 0 && threadIdx.x == 0) *loss_slot = 0.0f;
  int wid = threadIdx.x >> 5, lane = threadIdx.x & 31;
  int row = blockIdx.x * 8 + wid;
  const float4* src = (const float4*)(cb + (size_t)row * DC);
  float4 a = src[lane * 2 + 0];
  float4 b = src[lane * 2 + 1];
  float s = a.x * a.x + a.y * a.y + a.z * a.z + a.w * a.w +
            b.x * b.x + b.y * b.y + b.z * b.z + b.w * b.w;
#pragma unroll
  for (int off = 16; off >= 1; off >>= 1) s += __shfl_xor(s, off, 32);
  float inv = 1.0f / fmaxf(sqrtf(s), 1e-12f);
  float4 an = {a.x * inv, a.y * inv, a.z * inv, a.w * inv};
  float4 bn = {b.x * inv, b.y * inv, b.z * inv, b.w * inv};
  float4* dstf = (float4*)(cn_f32 + (size_t)row * DC);
  dstf[lane * 2 + 0] = an;
  dstf[lane * 2 + 1] = bn;
  union { __bf16 h[8]; int4 u; } p;
  p.h[0] = (__bf16)an.x; p.h[1] = (__bf16)an.y; p.h[2] = (__bf16)an.z; p.h[3] = (__bf16)an.w;
  p.h[4] = (__bf16)bn.x; p.h[5] = (__bf16)bn.y; p.h[6] = (__bf16)bn.z; p.h[7] = (__bf16)bn.w;
  ((int4*)(cn_b + (size_t)row * DC))[lane] = p.u;
}

// ---------------------------------------------------------------------------
// Kernel 2: f32 -> bf16 conversion of w_z [256x1024] and w_q [1024x256].
// ---------------------------------------------------------------------------
__global__ void __launch_bounds__(256)
cvt_bf16(const float* __restrict__ a, const float* __restrict__ b,
         __bf16* __restrict__ oa, __bf16* __restrict__ ob, int n) {
  int i = blockIdx.x * 256 + threadIdx.x;
  if (i < n)          oa[i]     = (__bf16)a[i];
  else if (i < 2 * n) ob[i - n] = (__bf16)b[i - n];
}

// ---------------------------------------------------------------------------
// Kernel 3: z_norm = l2norm(z @ w_z^T).  M-tile = 16 rows per block,
// N = 256 cols (8 waves x 2 tiles), K = 1024 (32 wmma steps).
// z tile staged to LDS as bf16 (32 KB); row sum-of-squares via LDS atomics.
// ---------------------------------------------------------------------------
__global__ void __launch_bounds__(256)
gemm1_norm(const float* __restrict__ z, const __bf16* __restrict__ wz,
           float* __restrict__ zn_f32, __bf16* __restrict__ zn_b) {
  __shared__ __align__(32) __bf16 sA[16][DHID];  // 32 KB
  __shared__ float rowsq[16];
  int tid = threadIdx.x, wid = tid >> 5, lane = tid & 31;
  int m = lane & 15, hi = lane >> 4;
  int row0 = blockIdx.x * 16;
  if (tid < 16) rowsq[tid] = 0.0f;
  {  // stage: 256 threads x 64 contiguous elems (f32 -> bf16 conversion)
    int r = tid >> 4, seg = tid & 15;
    const float4* src = (const float4*)(z + (size_t)(row0 + r) * DHID + seg * 64);
#pragma unroll
    for (int j = 0; j < 16; ++j) {
      float4 f = src[j];
      union { __bf16 h[4]; unsigned long long u; } p;
      p.h[0] = (__bf16)f.x; p.h[1] = (__bf16)f.y;
      p.h[2] = (__bf16)f.z; p.h[3] = (__bf16)f.w;
      *(unsigned long long*)&sA[r][seg * 64 + j * 4] = p.u;
    }
  }
  __syncthreads();

  int c0 = wid * 32, c1 = c0 + 16;
  v8f acc0 = {}, acc1 = {};
  for (int ks = 0; ks < 32; ++ks) {
    int kb = ks * 32;
    AFrag af;
    af.h[0] = *(const v8bf*)&sA[m][kb + hi * 16];
    af.h[1] = *(const v8bf*)&sA[m][kb + 8 + hi * 16];
    v16bf b0 = *(const v16bf*)(wz + (size_t)(c0 + m) * DHID + kb + hi * 16);
    v16bf b1 = *(const v16bf*)(wz + (size_t)(c1 + m) * DHID + kb + hi * 16);
    acc0 = wmma_bf16(af.v, b0, acc0);
    acc1 = wmma_bf16(af.v, b1, acc1);
  }
#pragma unroll
  for (int r = 0; r < 8; ++r)
    atomicAdd(&rowsq[r + 8 * hi], acc0[r] * acc0[r] + acc1[r] * acc1[r]);
  __syncthreads();
#pragma unroll
  for (int r = 0; r < 8; ++r) {
    int mr = r + 8 * hi;
    float inv = 1.0f / fmaxf(sqrtf(rowsq[mr]), 1e-12f);
    size_t base = (size_t)(row0 + mr) * DC;
    float v0 = acc0[r] * inv, v1 = acc1[r] * inv;
    zn_f32[base + c0 + m] = v0;
    zn_f32[base + c1 + m] = v1;
    zn_b[base + c0 + m] = (__bf16)v0;
    zn_b[base + c1 + m] = (__bf16)v1;
  }
}

// ---------------------------------------------------------------------------
// Kernel 4: codes = argmax_k (z_norm . code_norm[k]) (== argmin dist).
// M-tile = 128 rows/block (8 waves x 16-row subtiles); all waves share 8 KB
// code tiles double-buffered in LDS via async global->LDS DMA; A fragments
// resident (64 VGPRs/lane).
// ---------------------------------------------------------------------------
__global__ void __launch_bounds__(256)
dist_argmax(const __bf16* __restrict__ zn, const __bf16* __restrict__ cn,
            int* __restrict__ codes, float* __restrict__ codes_f) {
  __shared__ __align__(32) __bf16 sB[2][16][DC];  // 2 x 8 KB
  int tid = threadIdx.x, wid = tid >> 5, lane = tid & 31;
  int m = lane & 15, hi = lane >> 4;
  int row0 = blockIdx.x * 128 + wid * 16;

  AFrag af[8];
#pragma unroll
  for (int ks = 0; ks < 8; ++ks) {
    size_t base = (size_t)(row0 + m) * DC + ks * 32;
    af[ks].h[0] = *(const v8bf*)(zn + base + hi * 16);
    af[ks].h[1] = *(const v8bf*)(zn + base + 8 + hi * 16);
  }
  float bestv[8];
  int   besti[8];
#pragma unroll
  for (int r = 0; r < 8; ++r) { bestv[r] = -3.0e38f; besti[r] = 0; }

  int sr = tid >> 4, sseg = tid & 15;  // staging: 32 B per thread
  copy32B_to_lds(cn + (size_t)sr * DC + sseg * 16, &sB[0][sr][sseg * 16]);
  wait_async_lds();
  __syncthreads();

  const int NT = KCODE / 16;  // 512
  for (int nt = 0; nt < NT; ++nt) {
    int buf = nt & 1;
    if (nt + 1 < NT)  // fire DMA for the next tile into the other buffer
      copy32B_to_lds(cn + (size_t)((nt + 1) * 16 + sr) * DC + sseg * 16,
                     &sB[buf ^ 1][sr][sseg * 16]);
    if (nt + 2 < NT)
      __builtin_prefetch(cn + (size_t)((nt + 2) * 16 + sr) * DC + sseg * 16, 0, 0);

    v8f acc = {};
#pragma unroll
    for (int ks = 0; ks < 8; ++ks) {
      v16bf b = *(const v16bf*)&sB[buf][m][ks * 32 + hi * 16];
      acc = wmma_bf16(af[ks].v, b, acc);
    }
#pragma unroll
    for (int r = 0; r < 8; ++r) {
      if (acc[r] > bestv[r]) { bestv[r] = acc[r]; besti[r] = nt * 16 + m; }
    }
    wait_async_lds();
    __syncthreads();
  }

  // reduce best across the 16 N-lanes of each half-wave
#pragma unroll
  for (int r = 0; r < 8; ++r) {
    float v = bestv[r];
    int   i = besti[r];
#pragma unroll
    for (int off = 8; off >= 1; off >>= 1) {
      float ov = __shfl_xor(v, off, 16);
      int   oi = __shfl_xor(i, off, 16);
      if (ov > v || (ov == v && oi < i)) { v = ov; i = oi; }
    }
    if (m == 0) {
      int row = row0 + r + 8 * hi;
      codes[row] = i;
      codes_f[row] = (float)i;
    }
  }
}

// ---------------------------------------------------------------------------
// Kernel 5: q = code_norm[codes] @ w_q^T.  16 rows/block, 1024 cols
// (8 waves x 8 N-tiles), K = 256. Gathered A tile staged in LDS (8 KB) via
// async DMA (per-lane gather addresses).
// ---------------------------------------------------------------------------
__global__ void __launch_bounds__(256)
gather_gemm2(const int* __restrict__ codes, const __bf16* __restrict__ cn,
             const __bf16* __restrict__ wq, float* __restrict__ q) {
  __shared__ __align__(32) __bf16 sA[16][DC];  // 8 KB
  int tid = threadIdx.x, wid = tid >> 5, lane = tid & 31;
  int m = lane & 15, hi = lane >> 4;
  int row0 = blockIdx.x * 16;
  {
    int sr = tid >> 4, sseg = tid & 15;
    int code = codes[row0 + sr];
    copy32B_to_lds(cn + (size_t)code * DC + sseg * 16, &sA[sr][sseg * 16]);
  }
  wait_async_lds();
  __syncthreads();
  AFrag af[8];
#pragma unroll
  for (int ks = 0; ks < 8; ++ks) {
    af[ks].h[0] = *(const v8bf*)&sA[m][ks * 32 + hi * 16];
    af[ks].h[1] = *(const v8bf*)&sA[m][ks * 32 + 8 + hi * 16];
  }
  for (int j = 0; j < 8; ++j) {
    int c0 = (wid * 8 + j) * 16;
    v8f acc = {};
#pragma unroll
    for (int ks = 0; ks < 8; ++ks) {
      v16bf b = *(const v16bf*)(wq + (size_t)(c0 + m) * DC + ks * 32 + hi * 16);
      acc = wmma_bf16(af[ks].v, b, acc);
    }
#pragma unroll
    for (int r = 0; r < 8; ++r)
      q[(size_t)(row0 + r + 8 * hi) * DHID + c0 + m] = acc[r];
  }
}

// ---------------------------------------------------------------------------
// Kernel 6: quantize_loss = 1.25 * mean((q_norm - z_norm)^2)
// ---------------------------------------------------------------------------
__global__ void __launch_bounds__(256)
quant_loss(const int* __restrict__ codes, const float* __restrict__ cn,
           const float* __restrict__ zn, float* __restrict__ loss) {
  const long long N = (long long)MROWS * DC;
  float s = 0.0f;
  for (long long e = (long long)blockIdx.x * 256 + threadIdx.x; e < N;
       e += (long long)gridDim.x * 256) {
    int t = (int)(e >> 8), c = (int)(e & 255);
    float d = cn[(size_t)codes[t] * DC + c] - zn[e];
    s += d * d;
  }
#pragma unroll
  for (int off = 16; off >= 1; off >>= 1) s += __shfl_xor(s, off, 32);
  __shared__ float wsum[8];
  int wid = threadIdx.x >> 5, lane = threadIdx.x & 31;
  if (lane == 0) wsum[wid] = s;
  __syncthreads();
  if (threadIdx.x < 8) {
    float v = wsum[threadIdx.x];
#pragma unroll
    for (int off = 4; off >= 1; off >>= 1) v += __shfl_xor(v, off, 8);
    if (threadIdx.x == 0) atomicAdd(loss, v * (1.25f / (float)N));
  }
}

// ---------------------------------------------------------------------------
extern "C" void kernel_launch(void* const* d_in, const int* in_sizes, int n_in,
                              void* d_out, int out_size, void* d_ws, size_t ws_size,
                              hipStream_t stream) {
  const float* z  = (const float*)d_in[0];  // [8,2048,1024]
  const float* cb = (const float*)d_in[1];  // [8192,256]
  const float* wz = (const float*)d_in[2];  // [256,1024]
  const float* wq = (const float*)d_in[3];  // [1024,256]

  float* out     = (float*)d_out;
  float* q       = out;                 // 16,777,216 floats
  float* codes_f = out + 16777216;      // 16,384
  float* loss    = out + 16793600;      // 1

  char* ws = (char*)d_ws;
  float*  cn_f32 = (float*)(ws + 0);           //  8 MB
  float*  zn_f32 = (float*)(ws + 8388608);     // 16 MB
  __bf16* cn_b   = (__bf16*)(ws + 25165824);   //  4 MB
  __bf16* zn_b   = (__bf16*)(ws + 29360128);   //  8 MB
  __bf16* wz_b   = (__bf16*)(ws + 37748736);   // 512 KB
  __bf16* wq_b   = (__bf16*)(ws + 38273024);   // 512 KB
  int*    code_i = (int*)(ws + 38797312);      //  64 KB

  prep_codebook<<<1024, 256, 0, stream>>>(cb, cn_f32, cn_b, loss);
  cvt_bf16<<<2048, 256, 0, stream>>>(wz, wq, wz_b, wq_b, 256 * 1024);
  gemm1_norm<<<1024, 256, 0, stream>>>(z, wz_b, zn_f32, zn_b);
  dist_argmax<<<128, 256, 0, stream>>>(zn_b, cn_b, code_i, codes_f);
  gather_gemm2<<<1024, 256, 0, stream>>>(code_i, cn_b, wq_b, q);
  quant_loss<<<256, 256, 0, stream>>>(code_i, cn_f32, zn_f32, loss);
}
